// ESN_73521250173428
// MI455X (gfx1250) — matrix operand order repaired and checked by reading, not compiled, not run on hardware
//
#include <hip/hip_runtime.h>
#include <hip/hip_bf16.h>
#include <stdint.h>
#include <math.h>

// ---------------- problem constants ----------------
#define RES  1024
#define INP  16
#define OUTD 8
#define BATCH 32
#define SEQ  4096
#define LEAK 0.7f

// ---------------- vector types ----------------
typedef __bf16  bf16_t;
typedef bf16_t  v16bf __attribute__((ext_vector_type(16)));
typedef bf16_t  v8bf  __attribute__((ext_vector_type(8)));
typedef float   v8f   __attribute__((ext_vector_type(8)));

__device__ __forceinline__ unsigned short f2bf(float f) {
  unsigned u = __float_as_uint(f);
  unsigned r = 0x7FFFu + ((u >> 16) & 1u);
  return (unsigned short)((u + r) >> 16);
}

__device__ __forceinline__ v16bf cat8(v8bf lo, v8bf hi) {
  return __builtin_shufflevector(lo, hi, 0,1,2,3,4,5,6,7,8,9,10,11,12,13,14,15);
}

__device__ __forceinline__ v8f wmma_bf16(v16bf a, v16bf b, v8f c) {
  // v_wmma_f32_16x16x32_bf16
  return __builtin_amdgcn_wmma_f32_16x16x32_bf16(false, a, false, b, (short)0, c, false, false);
}

// ---------------- TDM (Tensor Data Mover) ----------------
#if defined(__has_builtin)
#  if __has_builtin(__builtin_amdgcn_tensor_load_to_lds)
#    define HAVE_TDM 1
#  endif
#endif
#if __has_include(<hip/amd_detail/amd_gfx1250_TDM.h>)
#  define TDM_SIX_ARGS 1
#endif

#if defined(HAVE_TDM)
typedef unsigned tdmu4 __attribute__((ext_vector_type(4)));
typedef int      tdmi8 __attribute__((ext_vector_type(8)));
typedef int      tdmi4 __attribute__((ext_vector_type(4)));

// 2D bf16 tile, full rows: tile_dim0 = tensor_dim0 = stride0 (elements of 2B)
__device__ __forceinline__ void tdm_load_2d(const void* g, unsigned lds_off,
                                            unsigned d0, unsigned d1, unsigned stride0) {
  unsigned long long ga = (unsigned long long)g;
  tdmu4 g0;
  g0[0] = 1u;                                            // count=1, user descriptor
  g0[1] = lds_off;                                       // LDS byte address
  g0[2] = (unsigned)(ga & 0xFFFFFFFFu);                  // global addr lo
  g0[3] = (unsigned)((ga >> 32) & 0x1FFFFFFu) | (2u << 30); // addr hi + type=2
  tdmi8 g1;
  g1[0] = (int)(1u << 16);                               // data_size=1 -> 2 bytes
  g1[1] = (int)((d0 & 0xFFFFu) << 16);                   // tensor_dim0 lo16 @ [63:48]
  g1[2] = (int)(((d0 >> 16) & 0xFFFFu) | ((d1 & 0xFFFFu) << 16)); // dim0 hi16 | dim1 lo16
  g1[3] = (int)(((d1 >> 16) & 0xFFFFu) | ((d0 & 0xFFFFu) << 16)); // dim1 hi16 | tile_dim0
  g1[4] = (int)(d1 & 0xFFFFu);                           // tile_dim1 (tile_dim2=0)
  g1[5] = (int)stride0;                                  // tensor_dim0_stride lo32
  g1[6] = 0;
  g1[7] = 0;
  tdmi4 gz = {0, 0, 0, 0};
#if defined(TDM_SIX_ARGS)
  tdmi8 gz8 = {0, 0, 0, 0, 0, 0, 0, 0};
  __builtin_amdgcn_tensor_load_to_lds(g0, g1, gz, gz, gz8, 0);
#else
  __builtin_amdgcn_tensor_load_to_lds(g0, g1, gz, gz, 0);
#endif
}
#endif // HAVE_TDM

// ---------------- prep kernels: pack weights into WMMA fragment layouts ----------------
// B-fragment layout (16-bit B 32x16): lane = half*16 + n  (half h holds K 8h..8h+7 and 16+8h..23+8h),
// 16 contiguous bf16 per lane -> element index ((tile*kc)*32 + lane)*16 + j

__global__ void prep_wres(const float* __restrict__ Wres, unsigned short* __restrict__ Wpack) {
  int idx  = blockIdx.x * blockDim.x + threadIdx.x;       // < 1048576
  int nt   = idx >> 14;
  int rem  = idx & 16383;
  int kc   = rem >> 9;
  int lane = (rem >> 4) & 31;
  int j    = rem & 15;
  int n    = nt * 16 + (lane & 15);
  int k    = kc * 32 + ((lane >> 4) << 3) + (j & 7) + ((j >> 3) << 4);
  Wpack[idx] = f2bf(Wres[n * RES + k]);                   // B[k][n] = W_res[n][k]
}

__global__ void prep_win(const float* __restrict__ Win, unsigned short* __restrict__ Winpack) {
  int idx  = blockIdx.x * blockDim.x + threadIdx.x;       // < 32768
  int nt   = idx >> 9;
  int lane = (idx >> 4) & 31;
  int j    = idx & 15;
  int n    = nt * 16 + (lane & 15);
  float v  = (j < 8) ? Win[n * INP + ((lane >> 4) << 3) + j] : 0.0f;       // K=16..31 zero pad
  Winpack[idx] = f2bf(v);
}

__global__ void prep_wout(const float* __restrict__ Wout, unsigned short* __restrict__ Woutpack) {
  int idx  = blockIdx.x * blockDim.x + threadIdx.x;       // < 16384
  int kc   = idx >> 9;
  int lane = (idx >> 4) & 31;
  int j    = idx & 15;
  int k    = kc * 32 + ((lane >> 4) << 3) + (j & 7) + ((j >> 3) << 4);
  int o    = lane & 15;
  float v  = (o < OUTD) ? Wout[o * RES + k] : 0.0f;       // N=8..15 zero pad
  Woutpack[idx] = f2bf(v);
}

__global__ void prep_x(const float* __restrict__ x, unsigned short* __restrict__ xb) {
  int idx = blockIdx.x * blockDim.x + threadIdx.x;        // < 2097152
  xb[idx] = f2bf(x[idx]);
}

__global__ void prep_h(const float* __restrict__ state, float* __restrict__ hf,
                       unsigned short* __restrict__ hbf, unsigned* __restrict__ counter) {
  int idx = blockIdx.x * blockDim.x + threadIdx.x;        // < 32768
  float v = state[idx];
  hf[idx]  = v;
  hbf[idx] = f2bf(v);                                     // buffer 0 = h_0
  if (idx == 0) *counter = 0u;
}

// ---------------- grid-wide barrier (16 resident workgroups) ----------------
__device__ __forceinline__ void grid_sync(unsigned* cnt, unsigned target) {
  __builtin_amdgcn_fence(__ATOMIC_RELEASE, "agent");
  __syncthreads();
  if (threadIdx.x == 0) {
    __hip_atomic_fetch_add(cnt, 1u, __ATOMIC_RELAXED, __HIP_MEMORY_SCOPE_AGENT);
    while (__hip_atomic_load(cnt, __ATOMIC_RELAXED, __HIP_MEMORY_SCOPE_AGENT) < target)
      __builtin_amdgcn_s_sleep(2);
  }
  __syncthreads();
  __builtin_amdgcn_fence(__ATOMIC_ACQUIRE, "agent");
}

// ---------------- persistent scan kernel ----------------
// 16 WGs x 256 threads (8 wave32). Wave = one 16x16 output tile of h.
// LDS: [0,128K) W_res B-fragments (4 N-tiles), [128K,132K) W_in frags, [132K,164K) W_out frags.
__global__ __launch_bounds__(256) void esn_scan(
    const unsigned short* __restrict__ Wpack,
    const unsigned short* __restrict__ Winpack,
    const unsigned short* __restrict__ Woutpack,
    const unsigned short* __restrict__ xb,
    unsigned short* __restrict__ hbf,      // [2][32][1024] bf16 double buffer
    float* __restrict__ hf,                // [32][1024] f32 leak state
    const float* __restrict__ outb,        // [8] bias
    float* __restrict__ out,               // [32][4096][8] f32 + [32][1024] final state
    unsigned* __restrict__ counter) {
  extern __shared__ char smem[];
  char* ldsW    = smem;             // 131072 B
  char* ldsWin  = smem + 131072;    //   4096 B
  char* ldsWout = smem + 135168;    //  32768 B

  const int tid  = threadIdx.x;
  const int lane = tid & 31;
  const int wave = tid >> 5;
  const int wg   = blockIdx.x;

  // ---- fill LDS with this WG's weight panels ----
#if defined(HAVE_TDM)
  if (wave == 0) {
    tdm_load_2d(Wpack + (size_t)wg * 65536, (unsigned)(size_t)ldsW, 16384u, 4u, 16384u);
    tdm_load_2d(Winpack + wg * 2048,        (unsigned)(size_t)ldsWin, 2048u, 1u, 2048u);
    tdm_load_2d(Woutpack,                   (unsigned)(size_t)ldsWout, 16384u, 1u, 16384u);
    __builtin_amdgcn_s_wait_tensorcnt(0);
  }
#else
  {
    const uint4* g0 = (const uint4*)(Wpack + (size_t)wg * 65536);
    uint4* l0 = (uint4*)ldsW;
    for (int i = tid; i < 8192; i += 256) l0[i] = g0[i];
    const uint4* g1 = (const uint4*)(Winpack + wg * 2048);
    uint4* l1 = (uint4*)ldsWin;
    for (int i = tid; i < 256; i += 256) l1[i] = g1[i];
    const uint4* g2 = (const uint4*)Woutpack;
    uint4* l2 = (uint4*)ldsWout;
    for (int i = tid; i < 2048; i += 256) l2[i] = g2[i];
  }
#endif
  __syncthreads();

  const int m     = wave >> 2;              // M tile: 0..1 (batch rows m*16..)
  const int ntL   = wave & 3;               // local N tile 0..3
  const int nt    = wg * 4 + ntL;           // global N tile 0..63
  const int m0    = m * 16;
  const int col   = nt * 16 + (lane & 15);  // this lane's C column (res index)
  const int half8 = (lane >> 4) << 3;       // A-frag K sub-offset: 0 or 8
  const int arow  = m0 + (lane & 15);       // A-frag row (batch) this lane loads
  const int mAdd  = (lane & 16) ? 8 : 0;    // C layout row offset

  const char* ldsB = ldsW + ntL * 32768 + lane * 32;
  const v16bf winB = *(const v16bf*)(ldsWin + ntL * 1024 + lane * 32);
  const unsigned nwg = gridDim.x;

  unsigned p = 0;
  for (int t = 0; t < SEQ; ++t) {
    const unsigned short* hr = hbf + p * (BATCH * RES);
    unsigned short*       hw = hbf + (p ^ 1) * (BATCH * RES);

    // input projection u_t: one WMMA, K padded 16->32 with zeros
    v8bf xlo = *(const v8bf*)(xb + ((size_t)arow * SEQ + t) * INP + half8);
    v8bf z8  = {};
    v8f acc  = wmma_bf16(cat8(xlo, z8), winB, (v8f){});

    // h @ W_res^T : 32 K-chunks of 32
    const unsigned short* hrow = hr + arow * RES + half8;
    #pragma unroll 4
    for (int kc = 0; kc < 32; ++kc) {
      v8bf alo = *(const v8bf*)(hrow + kc * 32);
      v8bf ahi = *(const v8bf*)(hrow + kc * 32 + 16);
      v16bf bb = *(const v16bf*)(ldsB + kc * 1024);
      acc = wmma_bf16(cat8(alo, ahi), bb, acc);
    }

    // leaky-tanh update, write f32 state + bf16 state (next step's A input)
    #pragma unroll
    for (int r = 0; r < 8; ++r) {
      int brow = m0 + mAdd + r;
      int idx  = brow * RES + col;
      float nh = (1.0f - LEAK) * hf[idx] + LEAK * tanhf(acc[r]);
      hf[idx]  = nh;
      hw[idx]  = f2bf(nh);
    }

    grid_sync(counter, nwg * (unsigned)(t + 1));

    // fused readout for step t (2 waves; overlaps others' step t+1)
    if (wg == 0 && ntL == 0) {
      v8f oacc = {};
      const unsigned short* srow = hw + arow * RES + half8;
      #pragma unroll 4
      for (int kc = 0; kc < 32; ++kc) {
        v8bf alo = *(const v8bf*)(srow + kc * 32);
        v8bf ahi = *(const v8bf*)(srow + kc * 32 + 16);
        v16bf bb = *(const v16bf*)(ldsWout + kc * 1024 + lane * 32);
        oacc = wmma_bf16(cat8(alo, ahi), bb, oacc);
      }
      int o = lane & 15;
      if (o < OUTD) {
        float bias = outb[o];
        #pragma unroll
        for (int r = 0; r < 8; ++r) {
          int brow = m0 + mAdd + r;
          out[((size_t)brow * SEQ + t) * OUTD + o] = oacc[r] + bias;
        }
      }
    }
    p ^= 1;
  }

  // final state (f32), appended after outputs
  float* fs = out + (size_t)BATCH * SEQ * OUTD;
  #pragma unroll
  for (int r = 0; r < 8; ++r) {
    int idx = (m0 + mAdd + r) * RES + col;
    fs[idx] = hf[idx];
  }
}

// ---------------- launch ----------------
extern "C" void kernel_launch(void* const* d_in, const int* in_sizes, int n_in,
                              void* d_out, int out_size, void* d_ws, size_t ws_size,
                              hipStream_t stream) {
  (void)in_sizes; (void)n_in; (void)out_size; (void)ws_size;
  const float* x       = (const float*)d_in[0];
  const float* state   = (const float*)d_in[1];
  const float* W_in    = (const float*)d_in[2];
  const float* W_res   = (const float*)d_in[3];
  const float* W_out_w = (const float*)d_in[4];
  const float* W_out_b = (const float*)d_in[5];

  char* ws = (char*)d_ws;
  unsigned short* Wpack    = (unsigned short*)(ws);                         // 2 MB
  unsigned short* xb       = (unsigned short*)(ws + (size_t)(2 << 20));     // 4 MB
  size_t o = (size_t)(6 << 20);
  unsigned short* Winpack  = (unsigned short*)(ws + o);          o += 65536;  // 64 KB
  unsigned short* Woutpack = (unsigned short*)(ws + o);          o += 32768;  // 32 KB
  unsigned short* hbf      = (unsigned short*)(ws + o);          o += 131072; // 128 KB (x2 buf)
  float*          hf       = (float*)(ws + o);                   o += 131072; // 128 KB
  unsigned*       counter  = (unsigned*)(ws + o);

  prep_wres<<<4096, 256, 0, stream>>>(W_res, Wpack);
  prep_win <<<128,  256, 0, stream>>>(W_in, Winpack);
  prep_wout<<<64,   256, 0, stream>>>(W_out_w, Woutpack);
  prep_x   <<<8192, 256, 0, stream>>>(x, xb);
  prep_h   <<<128,  256, 0, stream>>>(state, hf, hbf, counter);

  esn_scan<<<16, 256, 167936, stream>>>(Wpack, Winpack, Woutpack, xb, hbf, hf,
                                        W_out_b, (float*)d_out, counter);
}
